// TransformerDecoderLayer_4810363372429
// MI455X (gfx1250) — matrix operand order repaired and checked by reading, not compiled, not run on hardware
//
#include <hip/hip_runtime.h>

// ---------- types ----------
typedef __bf16 bf16x16 __attribute__((ext_vector_type(16)));
typedef float  v8f     __attribute__((ext_vector_type(8)));
typedef unsigned int u32x4 __attribute__((ext_vector_type(4)));
typedef int          i32x8 __attribute__((ext_vector_type(8)));
typedef int          i32x4 __attribute__((ext_vector_type(4)));

union Frag {
    bf16x16        bf;
    uint4          q[2];
    unsigned short us[16];
};

// Tensor Data Mover availability (this toolchain: 6-arg builtin; fallback = cooperative load)
#if defined(__HIP_DEVICE_COMPILE__) && defined(__gfx1250__)
#if __has_builtin(__builtin_amdgcn_tensor_load_to_lds) && __has_builtin(__builtin_amdgcn_s_wait_tensorcnt)
#define USE_TDM 1
#endif
#endif
#ifndef USE_TDM
#define USE_TDM 0
#endif

__device__ __forceinline__ unsigned short f2bf(float f) {
    unsigned int u = __float_as_uint(f);
    unsigned int r = u + 0x7FFFu + ((u >> 16) & 1u);   // round-to-nearest-even
    return (unsigned short)(r >> 16);
}

// ---------- fp32 -> bf16 convert ----------
__global__ void cvt_f32_bf16(const float* __restrict__ in, unsigned short* __restrict__ out, int n) {
    int i = blockIdx.x * blockDim.x + threadIdx.x;
    int stride = gridDim.x * blockDim.x;
    for (; i < n; i += stride) out[i] = f2bf(in[i]);
}

// ---------- LayerNorm (1024 cols), fp32 in -> bf16 out ----------
__device__ __forceinline__ float blockReduceSum(float v, float* red) {
    for (int o = 16; o > 0; o >>= 1) v += __shfl_down(v, o, 32);
    int wid = threadIdx.x >> 5;
    if ((threadIdx.x & 31) == 0) red[wid] = v;
    __syncthreads();
    float tot = 0.f;
    if (threadIdx.x < 8) tot = red[threadIdx.x];
    tot += __shfl_down(tot, 4, 32);
    tot += __shfl_down(tot, 2, 32);
    tot += __shfl_down(tot, 1, 32);
    if (threadIdx.x == 0) red[0] = tot;
    __syncthreads();
    float r = red[0];
    __syncthreads();
    return r;
}

__global__ __launch_bounds__(256) void ln_bf16(const float* __restrict__ x,
                                               const float* __restrict__ g,
                                               const float* __restrict__ bb,
                                               unsigned short* __restrict__ o) {
    __shared__ float red[8];
    int row = blockIdx.x, t = threadIdx.x;
    const float* xr = x + (size_t)row * 1024;
    float v[4];
    float s = 0.f;
#pragma unroll
    for (int j = 0; j < 4; j++) { v[j] = xr[t + j * 256]; s += v[j]; }
    float mean = blockReduceSum(s, red) * (1.f / 1024.f);
    float s2 = 0.f;
#pragma unroll
    for (int j = 0; j < 4; j++) { float d = v[j] - mean; s2 += d * d; }
    float var = blockReduceSum(s2, red) * (1.f / 1024.f);
    float inv = rsqrtf(var + 1e-6f);
    unsigned short* orow = o + (size_t)row * 1024;
#pragma unroll
    for (int j = 0; j < 4; j++) {
        int c = t + j * 256;
        orow[c] = f2bf((v[j] - mean) * inv * g[c] + bb[c]);
    }
}

// ---------- GEMM: out[M,N] = act( A_bf16[M,K] @ W_bf16[K,N] + bias ) (+ resid) ----------
// Block: 256 threads (8 waves). Tile 128x64, BK=32. Wave w owns rows w*16..w*16+15, all 64 cols.
template <bool GELU, bool RESID, bool OUTF32>
__global__ __launch_bounds__(256) void gemm_bf16(const unsigned short* __restrict__ A,
                                                 const unsigned short* __restrict__ W,
                                                 const float* __restrict__ bias,
                                                 const float* __restrict__ resid,
                                                 void* __restrict__ outp,
                                                 int Mtot, int N, int K) {
    __shared__ __align__(16) unsigned short lA[128 * 32];   // [m][k]
    __shared__ __align__(16) unsigned short lB[64 * 32];    // [n][k]  (transposed)
    int t = threadIdx.x;
    int n0 = blockIdx.x * 64, m0 = blockIdx.y * 128;
    int lane = t & 31, half = lane >> 4, lr = lane & 15;
    int wave = t >> 5;

    v8f acc[4] = {};

    int brow = t >> 3, bcol = (t & 7) * 8;   // B loader: 32 rows x 8 groups of 8
#if !USE_TDM
    int arow = t >> 1, acol = (t & 1) * 16;  // A loader: 128 rows x 2 groups of 16
#endif

    for (int k0 = 0; k0 < K; k0 += 32) {
#if USE_TDM
        if (t < 32) {   // wave 0 issues one TDM DMA for the whole 128x32 A tile
            unsigned long long ga = (unsigned long long)(const void*)A +
                                    ((unsigned long long)((size_t)m0 * K + k0) << 1);
            u32x4 g0;
            g0[0] = 1u;                                   // count=1 (valid descriptor)
            g0[1] = (unsigned)(__SIZE_TYPE__)(void*)lA;   // lds_addr (low 32 bits = LDS offset)
            g0[2] = (unsigned)(ga & 0xFFFFFFFFu);
            g0[3] = (unsigned)((ga >> 32) & 0x01FFFFFFu) | 0x80000000u;  // addr[56:32] | type=2
            i32x8 g1;
            g1[0] = (int)(1u << 16);                                        // data_size=1 -> 2B
            g1[1] = (int)(((unsigned)K & 0xFFFFu) << 16);                   // tensor_dim0[15:0]
            g1[2] = (int)((((unsigned)K >> 16) & 0xFFFFu) |
                          (((unsigned)Mtot & 0xFFFFu) << 16));              // dim0 hi | dim1 lo
            g1[3] = (int)((((unsigned)Mtot >> 16) & 0xFFFFu) | (32u << 16));// dim1 hi | tile_dim0=32
            g1[4] = (int)128u;                                              // tile_dim1=128
            g1[5] = (int)K;                                                 // tensor_dim0_stride
            g1[6] = 0;
            g1[7] = 0;
            i32x4 gz4 = {0, 0, 0, 0};
            i32x8 gz8 = {0, 0, 0, 0, 0, 0, 0, 0};
            __builtin_amdgcn_tensor_load_to_lds(g0, g1, gz4, gz4, gz8, 0);
            __builtin_amdgcn_s_wait_tensorcnt(0);
        }
#else
        *(uint4*)(lA + arow * 32 + acol) =
            *(const uint4*)(A + (size_t)(m0 + arow) * K + k0 + acol);
        *(uint4*)(lA + arow * 32 + acol + 8) =
            *(const uint4*)(A + (size_t)(m0 + arow) * K + k0 + acol + 8);
#endif
        // B tile: 32x64 loaded row-major, stored transposed [n][k]
        {
            union { uint4 q; unsigned short s[8]; } bu;
            bu.q = *(const uint4*)(W + (size_t)(k0 + brow) * N + n0 + bcol);
#pragma unroll
            for (int i = 0; i < 8; i++) lB[(bcol + i) * 32 + brow] = bu.s[i];
        }
        __syncthreads();

        Frag a;
        int ar = wave * 16 + lr;
        a.q[0] = *(const uint4*)(lA + ar * 32 + half * 8);        // K = half*8 .. +7
        a.q[1] = *(const uint4*)(lA + ar * 32 + 16 + half * 8);   // K = 16+half*8 .. +7
#pragma unroll
        for (int nt = 0; nt < 4; nt++) {
            Frag b;
            int nb = nt * 16 + lr;
            b.q[0] = *(const uint4*)(lB + nb * 32 + half * 8);
            b.q[1] = *(const uint4*)(lB + nb * 32 + 16 + half * 8);
            acc[nt] = __builtin_amdgcn_wmma_f32_16x16x32_bf16(false, a.bf, false, b.bf,
                                                              (short)0, acc[nt], false, false);
        }
        __syncthreads();
    }

#pragma unroll
    for (int nt = 0; nt < 4; nt++) {
        int n = n0 + nt * 16 + lr;
        float bv = bias ? bias[n] : 0.f;
#pragma unroll
        for (int r = 0; r < 8; r++) {
            int m = m0 + wave * 16 + half * 8 + r;
            float v = acc[nt][r] + bv;
            if (GELU) v = 0.5f * v * (1.f + erff(v * 0.70710678118654752f));
            if (RESID) v += resid[(size_t)m * N + n];
            if (OUTF32) ((float*)outp)[(size_t)m * N + n] = v;
            else        ((unsigned short*)outp)[(size_t)m * N + n] = f2bf(v);
        }
    }
    (void)Mtot;
}

// ---------- Flash attention ----------
// Block = 4 waves sharing (batch, head); wave w owns query tile mt = mtg*4+w.
// K chunk staged row-major in LDS, V chunk staged transposed; H=16, DH=64 hard-coded.
__global__ __launch_bounds__(128) void attn_bf16(const unsigned short* __restrict__ qp,
                                                 const unsigned short* __restrict__ kp,
                                                 const unsigned short* __restrict__ vp,
                                                 unsigned short* __restrict__ op,
                                                 int Lq, int Skv,
                                                 int sq, int sk, int sv, int causal) {
    __shared__ __align__(16) unsigned short ldsK[32 * 64];    // [key][dh]
    __shared__ __align__(16) unsigned short ldsVt[64 * 32];   // [dh][key]
    __shared__ __align__(16) unsigned short ldsP[4][16 * 32]; // per-wave P relayout patch
    int t = threadIdx.x;
    int wave = t >> 5, lane = t & 31, half = lane >> 4, lr = lane & 15;
    int ltg = (Lq >> 4) >> 2;                 // groups of 4 query tiles
    int mtg = blockIdx.x % ltg;
    int hh  = (blockIdx.x / ltg) & 15;
    int b   = blockIdx.x / (ltg * 16);
    int qbase = (mtg * 4 + wave) * 16;

    // Q fragments: K (=DH) 0..31 and 32..63
    const unsigned short* qrow = qp + (size_t)(b * Lq + qbase + lr) * sq + hh * 64;
    Frag qa0, qa1;
    qa0.q[0] = *(const uint4*)(qrow + half * 8);
    qa0.q[1] = *(const uint4*)(qrow + 16 + half * 8);
    qa1.q[0] = *(const uint4*)(qrow + 32 + half * 8);
    qa1.q[1] = *(const uint4*)(qrow + 48 + half * 8);

    float mrun[8], lrun[8];
#pragma unroll
    for (int r = 0; r < 8; r++) { mrun[r] = -1e30f; lrun[r] = 0.f; }
    v8f oacc[4] = {};

    int jmax_own = causal ? ((((qbase + 15) >> 5) + 1) << 5) : Skv;
    if (jmax_own > Skv) jmax_own = Skv;
    int jmax_blk = causal ? (((((mtg * 4 + 3) * 16 + 15) >> 5) + 1) << 5) : Skv;
    if (jmax_blk > Skv) jmax_blk = Skv;

    // staging index precompute
    int kr = t >> 2, kc = (t & 3) * 16;       // K: 32 rows x 4 groups of 16
    int vr = (t >> 3) * 2, vc = (t & 7) * 8;  // V: 16 row-pairs x 8 groups of 8

    for (int j0 = 0; j0 < jmax_blk; j0 += 32) {
        // ---- cooperative staging ----
        {
            const unsigned short* gk = kp + (size_t)(b * Skv + j0 + kr) * sk + hh * 64 + kc;
            *(uint4*)(ldsK + kr * 64 + kc)     = *(const uint4*)gk;
            *(uint4*)(ldsK + kr * 64 + kc + 8) = *(const uint4*)(gk + 8);
        }
        {
            const unsigned short* gv0 = vp + (size_t)(b * Skv + j0 + vr) * sv + hh * 64 + vc;
            union { uint4 q; unsigned short s[8]; } r0, r1;
            r0.q = *(const uint4*)gv0;
            r1.q = *(const uint4*)(gv0 + sv);
#pragma unroll
            for (int i = 0; i < 8; i++) {
                unsigned int pk = (unsigned int)r0.s[i] | ((unsigned int)r1.s[i] << 16);
                *(unsigned int*)(ldsVt + (vc + i) * 32 + vr) = pk;
            }
        }
        __syncthreads();

        if (j0 < jmax_own) {
            // ---- scores: Q(16x64) @ K^T(64x32) ----
            v8f s[2] = {};
#pragma unroll
            for (int jt = 0; jt < 2; jt++) {
                const unsigned short* krow = ldsK + (jt * 16 + lr) * 64;
                Frag kb;
                kb.q[0] = *(const uint4*)(krow + half * 8);
                kb.q[1] = *(const uint4*)(krow + 16 + half * 8);
                s[jt] = __builtin_amdgcn_wmma_f32_16x16x32_bf16(false, qa0.bf, false, kb.bf,
                                                                (short)0, s[jt], false, false);
                kb.q[0] = *(const uint4*)(krow + 32 + half * 8);
                kb.q[1] = *(const uint4*)(krow + 48 + half * 8);
                s[jt] = __builtin_amdgcn_wmma_f32_16x16x32_bf16(false, qa1.bf, false, kb.bf,
                                                                (short)0, s[jt], false, false);
            }

            // ---- online softmax ----
            float p0a[8], p1a[8], alpha[8];
#pragma unroll
            for (int r = 0; r < 8; r++) {
                int qi = qbase + half * 8 + r;
                float v0 = s[0][r] * 0.125f;
                float v1 = s[1][r] * 0.125f;
                if (causal) {
                    if (j0 + lr > qi)      v0 = -1e30f;
                    if (j0 + 16 + lr > qi) v1 = -1e30f;
                }
                float mv = fmaxf(v0, v1);
                mv = fmaxf(mv, __shfl_xor(mv, 1, 32));
                mv = fmaxf(mv, __shfl_xor(mv, 2, 32));
                mv = fmaxf(mv, __shfl_xor(mv, 4, 32));
                mv = fmaxf(mv, __shfl_xor(mv, 8, 32));
                float mn = fmaxf(mrun[r], mv);
                alpha[r] = __expf(mrun[r] - mn);
                float p0 = __expf(v0 - mn);
                float p1 = __expf(v1 - mn);
                float rs = p0 + p1;
                rs += __shfl_xor(rs, 1, 32);
                rs += __shfl_xor(rs, 2, 32);
                rs += __shfl_xor(rs, 4, 32);
                rs += __shfl_xor(rs, 8, 32);
                lrun[r] = lrun[r] * alpha[r] + rs;
                mrun[r] = mn;
                p0a[r] = p0; p1a[r] = p1;
            }

            // ---- relayout P (C/D layout -> A layout) via per-wave LDS patch ----
            unsigned short* lp = ldsP[wave];
#pragma unroll
            for (int r = 0; r < 8; r++) {
                int row = half * 8 + r;
                lp[row * 32 + lr]      = f2bf(p0a[r]);
                lp[row * 32 + 16 + lr] = f2bf(p1a[r]);
            }
            Frag pa;
            pa.q[0] = *(const uint4*)(lp + lr * 32 + half * 8);
            pa.q[1] = *(const uint4*)(lp + lr * 32 + 16 + half * 8);

#pragma unroll
            for (int nt = 0; nt < 4; nt++)
#pragma unroll
                for (int r = 0; r < 8; r++) oacc[nt][r] *= alpha[r];

            // ---- P(16x32) @ V(32x64) from transposed LDS ----
#pragma unroll
            for (int nt = 0; nt < 4; nt++) {
                const unsigned short* vrow = ldsVt + (nt * 16 + lr) * 32;
                Frag vb;
                vb.q[0] = *(const uint4*)(vrow + half * 8);
                vb.q[1] = *(const uint4*)(vrow + 16 + half * 8);
                oacc[nt] = __builtin_amdgcn_wmma_f32_16x16x32_bf16(false, pa.bf, false, vb.bf,
                                                                   (short)0, oacc[nt], false, false);
            }
        }
        __syncthreads();
    }

#pragma unroll
    for (int nt = 0; nt < 4; nt++)
#pragma unroll
        for (int r = 0; r < 8; r++) {
            int m = b * Lq + qbase + half * 8 + r;
            op[(size_t)m * 1024 + hh * 64 + nt * 16 + lr] = f2bf(oacc[nt][r] / lrun[r]);
        }
}

// ---------- host ----------
extern "C" void kernel_launch(void* const* d_in, const int* in_sizes, int n_in,
                              void* d_out, int out_size, void* d_ws, size_t ws_size,
                              hipStream_t stream) {
    (void)in_sizes; (void)n_in; (void)out_size; (void)ws_size;
    const float* x0        = (const float*)d_in[0];
    const float* src       = (const float*)d_in[1];
    const float* ln_self_w = (const float*)d_in[4];
    const float* ln_self_b = (const float*)d_in[5];
    const float* w_qkv     = (const float*)d_in[6];
    const float* b_qkv     = (const float*)d_in[7];
    const float* w_so      = (const float*)d_in[8];
    const float* b_so      = (const float*)d_in[9];
    const float* ln_cr_w   = (const float*)d_in[10];
    const float* ln_cr_b   = (const float*)d_in[11];
    const float* w_q       = (const float*)d_in[12];
    const float* b_q       = (const float*)d_in[13];
    const float* w_kv      = (const float*)d_in[14];
    const float* b_kv      = (const float*)d_in[15];
    const float* w_co      = (const float*)d_in[16];
    const float* b_co      = (const float*)d_in[17];
    const float* ln_ml_w   = (const float*)d_in[18];
    const float* ln_ml_b   = (const float*)d_in[19];
    const float* w_ff1     = (const float*)d_in[20];
    const float* b_ff1     = (const float*)d_in[21];
    const float* w_ff2     = (const float*)d_in[22];
    const float* b_ff2     = (const float*)d_in[23];

    char* p = (char*)d_ws;
    auto carve = [&](size_t bytes) -> char* {
        char* r = p;
        p += (bytes + 255) & ~(size_t)255;
        return r;
    };
    unsigned short* wqkvb = (unsigned short*)carve((size_t)3145728 * 2);
    unsigned short* wsob  = (unsigned short*)carve((size_t)1048576 * 2);
    unsigned short* wqb   = (unsigned short*)carve((size_t)1048576 * 2);
    unsigned short* wkvb  = (unsigned short*)carve((size_t)2097152 * 2);
    unsigned short* wcob  = (unsigned short*)carve((size_t)1048576 * 2);
    unsigned short* wff1b = (unsigned short*)carve((size_t)4194304 * 2);
    unsigned short* wff2b = (unsigned short*)carve((size_t)4194304 * 2);
    unsigned short* srcb  = (unsigned short*)carve((size_t)4194304 * 2);
    unsigned short* hb    = (unsigned short*)carve((size_t)4194304 * 2);
    unsigned short* big   = (unsigned short*)carve((size_t)16777216 * 2); // qkv then ff1
    unsigned short* kvb   = (unsigned short*)carve((size_t)8388608 * 2);
    unsigned short* a1    = (unsigned short*)carve((size_t)4194304 * 2);  // self-attn out, then q
    unsigned short* a2    = (unsigned short*)carve((size_t)4194304 * 2);  // cross-attn out
    float* x1 = (float*)carve((size_t)4194304 * 4);
    float* x2 = (float*)carve((size_t)4194304 * 4);

    // weight + source conversion
    cvt_f32_bf16<<<4096, 256, 0, stream>>>(w_qkv, wqkvb, 3145728);
    cvt_f32_bf16<<<4096, 256, 0, stream>>>(w_so,  wsob,  1048576);
    cvt_f32_bf16<<<4096, 256, 0, stream>>>(w_q,   wqb,   1048576);
    cvt_f32_bf16<<<4096, 256, 0, stream>>>(w_kv,  wkvb,  2097152);
    cvt_f32_bf16<<<4096, 256, 0, stream>>>(w_co,  wcob,  1048576);
    cvt_f32_bf16<<<4096, 256, 0, stream>>>(w_ff1, wff1b, 4194304);
    cvt_f32_bf16<<<4096, 256, 0, stream>>>(w_ff2, wff2b, 4194304);
    cvt_f32_bf16<<<4096, 256, 0, stream>>>(src,   srcb,  4194304);

    // ---- self-attention block ----
    ln_bf16<<<4096, 256, 0, stream>>>(x0, ln_self_w, ln_self_b, hb);
    gemm_bf16<false, false, false><<<dim3(48, 32), 256, 0, stream>>>(hb, wqkvb, b_qkv, nullptr, big, 4096, 3072, 1024);
    attn_bf16<<<1024, 128, 0, stream>>>(big, big + 1024, big + 2048, a1, 2048, 2048, 3072, 3072, 3072, 1);
    gemm_bf16<false, true, true><<<dim3(16, 32), 256, 0, stream>>>(a1, wsob, b_so, x0, x1, 4096, 1024, 1024);

    // ---- cross-attention block ----
    ln_bf16<<<4096, 256, 0, stream>>>(x1, ln_cr_w, ln_cr_b, hb);
    gemm_bf16<false, false, false><<<dim3(16, 32), 256, 0, stream>>>(hb, wqb, b_q, nullptr, a1, 4096, 1024, 1024);
    gemm_bf16<false, false, false><<<dim3(32, 32), 256, 0, stream>>>(srcb, wkvb, b_kv, nullptr, kvb, 4096, 2048, 1024);
    attn_bf16<<<1024, 128, 0, stream>>>(a1, kvb, kvb + 1024, a2, 2048, 2048, 1024, 2048, 2048, 0);
    gemm_bf16<false, true, true><<<dim3(16, 32), 256, 0, stream>>>(a2, wcob, b_co, x1, x2, 4096, 1024, 1024);

    // ---- MLP block ----
    ln_bf16<<<4096, 256, 0, stream>>>(x2, ln_ml_w, ln_ml_b, hb);
    gemm_bf16<true, false, false><<<dim3(64, 32), 256, 0, stream>>>(hb, wff1b, b_ff1, nullptr, big, 4096, 4096, 1024);
    gemm_bf16<false, true, true><<<dim3(16, 32), 256, 0, stream>>>(big, wff2b, b_ff2, x2, (float*)d_out, 4096, 1024, 4096);
}
